// FlaxGrok1BlocKSparesTop2MLPCollection_77429670412626
// MI455X (gfx1250) — compile-verified, matrix operand order checked by reading, not run on hardware
//
#include <hip/hip_runtime.h>
#include <hip/hip_bf16.h>
#include <math.h>

// ---------------------------------------------------------------------------
// Dense top-2 MoE (Grok1-style), 8 experts, B=4 S=2048 H=1024 I=4096.
// Compute-bound (~3.3 kFLOP/byte) -> bf16 WMMA (v_wmma_f32_16x16x32_bf16)
// with fp32 accumulation. Per expert (stream-serialized):
//   stage1: act = gelu_tanh(X@W1e) * (X@Wve)   (bf16 into d_ws, token-chunked)
//   stage2: out += w_e(token) * (act @ W2e)    (fp32 read-modify-write)
// Stage-2 A tile is staged with global_load_async_to_lds_b128 (ASYNCcnt).
// ---------------------------------------------------------------------------

typedef __bf16 bf16_t;
typedef __attribute__((ext_vector_type(16))) __bf16 v16bf;
typedef __attribute__((ext_vector_type(8)))  float  v8f;

#define DIM_M_TOTAL 8192   // B*S tokens
#define DIM_H       1024   // hidden
#define DIM_I       4096   // intermediate
#define NUM_EXPERTS 8
#define M_CHUNK     1024   // act chunk = 1024*4096*2B = 8 MB of d_ws

#define BM 128
#define BN 64
#define BK 32              // == WMMA K for bf16

// float -> bf16 via native hardware convert (v_cvt_bf16_f32 / v_cvt_pk_bf16_f32)
static __device__ __forceinline__ unsigned short f2bf(float f) {
    return __builtin_bit_cast(unsigned short, (__bf16)f);
}
static __device__ __forceinline__ unsigned pack2bf(float a, float b) {
    return (unsigned)f2bf(a) | ((unsigned)f2bf(b) << 16);
}

static __device__ __forceinline__ float gelu_tanh(float x) {
    const float c0 = 0.7978845608028654f, c1 = 0.044715f;
    float arg = c0 * (x + c1 * x * x * x);
#if __has_builtin(__builtin_amdgcn_tanhf)
    float t = __builtin_amdgcn_tanhf(arg);   // v_tanh_f32 (gfx1250 TRANS op)
#else
    float t = tanhf(arg);
#endif
    return 0.5f * x * (1.0f + t);
}

static __device__ __forceinline__ void wait_asynccnt0() {
#if __has_builtin(__builtin_amdgcn_s_wait_asynccnt)
    __builtin_amdgcn_s_wait_asynccnt(0);
#else
    asm volatile("s_wait_asynccnt 0x0" ::: "memory");
#endif
}

// Async bf16 global->LDS copy of 16B per lane (gfx1250 GLOBAL_LOAD_ASYNC_TO_LDS).
// lds_off: 32-bit LDS byte address (low 32 bits of generic shared pointer).
static __device__ __forceinline__ void async_copy_b128(unsigned lds_off,
                                                       const void* gaddr) {
    asm volatile("global_load_async_to_lds_b128 %0, %1, off"
                 :: "v"(lds_off), "v"(gaddr)
                 : "memory");
}

// Load one 16x32 bf16 WMMA operand fragment from an LDS tile laid out
// row-major [rows][32] (A: rows are M; B: rows are N, i.e. B^T).
// ISA 7.12.2 striping: lanes 0-15 hold K {0..7, 16..23}; lanes 16-31 hold
// K {8..15, 24..31} -> two contiguous 16B chunks per lane (ds_load_b128 x2).
static __device__ __forceinline__ v16bf load_frag(const unsigned short* lds,
                                                  int rowBase, int lane) {
    int r = lane & 15;
    int half = (lane >> 4) & 1;
    const unsigned short* base = lds + (rowBase + r) * BK + half * 8;
    union { uint4 q[2]; v16bf v; } u;
    u.q[0] = *(const uint4*)(base);
    u.q[1] = *(const uint4*)(base + 16);
    return u.v;
}

static __device__ __forceinline__ v8f wmma_bf16(v16bf a, v16bf b, v8f c) {
    // (neg_a, A, neg_b, B, c_mod, C, reuse_a, reuse_b)
    return __builtin_amdgcn_wmma_f32_16x16x32_bf16(false, a, false, b,
                                                   (short)0, c, false, false);
}

// ---------------------------------------------------------------------------
__global__ __launch_bounds__(256) void moe_zero_out(float4* out, int n4) {
    int i = blockIdx.x * 256 + threadIdx.x;
    if (i < n4) out[i] = make_float4(0.f, 0.f, 0.f, 0.f);
}

// ---------------------------------------------------------------------------
// Stage 1: one expert, one M-chunk.
//   X [M_CHUNK][H] f32 | W1,Wv [H][I] f32 | act [M_CHUNK][I] bf16 out
// grid = (M_CHUNK/BM, I/BN), block = 256 (8 waves, 4x2; 32x32 per wave).
__global__ __launch_bounds__(256) void moe_stage1(
        const float* __restrict__ X,
        const float* __restrict__ W1,
        const float* __restrict__ Wv,
        unsigned short* __restrict__ act) {
    __shared__ unsigned short sA[BM * BK];   // A tile, row-major [m][k]
    __shared__ unsigned short sG[BN * BK];   // W1 tile as B^T [n][k]
    __shared__ unsigned short sU[BN * BK];   // Wv tile as B^T [n][k]

    const int tid  = threadIdx.x;
    const int lane = tid & 31;
    const int wave = tid >> 5;
    const int wm = wave & 3;                 // 4 waves along M
    const int wn = wave >> 2;                // 2 waves along N
    const int m0 = blockIdx.x * BM;
    const int n0 = blockIdx.y * BN;

    v8f accG[2][2] = {};
    v8f accU[2][2] = {};

    // A staging: coalesced float4, 4 passes of 32 rows
    const int ca = (tid & 7) * 4;            // k offset
    const int ra = tid >> 3;                 // row 0..31
    // W staging: coalesced float4 along n, transposed scatter into LDS
    const int kw = tid >> 4;                 // k 0..15 (+16 second pass)
    const int nw = (tid & 15) * 4;           // n

    for (int k0 = 0; k0 < DIM_H; k0 += BK) {
        __syncthreads();   // previous iteration's fragment reads done

        // ---- issue ALL global loads first (latency overlap) ----
        float4 fa[4], fg[2], fu[2];
        #pragma unroll
        for (int p = 0; p < 4; ++p)
            fa[p] = *(const float4*)(X + (size_t)(m0 + ra + p * 32) * DIM_H
                                       + k0 + ca);
        #pragma unroll
        for (int p = 0; p < 2; ++p) {
            size_t off = (size_t)(k0 + kw + p * 16) * DIM_I + n0 + nw;
            fg[p] = *(const float4*)(W1 + off);
            fu[p] = *(const float4*)(Wv + off);
        }

        // ---- convert + store to LDS ----
        #pragma unroll
        for (int p = 0; p < 4; ++p) {
            uint2 d;
            d.x = pack2bf(fa[p].x, fa[p].y);
            d.y = pack2bf(fa[p].z, fa[p].w);
            *(uint2*)&sA[(ra + p * 32) * BK + ca] = d;   // ds_store_b64
        }
        #pragma unroll
        for (int p = 0; p < 2; ++p) {
            int k = kw + p * 16;
            sG[(nw + 0) * BK + k] = f2bf(fg[p].x);
            sG[(nw + 1) * BK + k] = f2bf(fg[p].y);
            sG[(nw + 2) * BK + k] = f2bf(fg[p].z);
            sG[(nw + 3) * BK + k] = f2bf(fg[p].w);
            sU[(nw + 0) * BK + k] = f2bf(fu[p].x);
            sU[(nw + 1) * BK + k] = f2bf(fu[p].y);
            sU[(nw + 2) * BK + k] = f2bf(fu[p].z);
            sU[(nw + 3) * BK + k] = f2bf(fu[p].w);
        }
        __syncthreads();

        // prefetch next K-tile of A into cache (global_prefetch_b8)
        if (k0 + BK < DIM_H)
            __builtin_prefetch(X + (size_t)(m0 + ra) * DIM_H + k0 + BK, 0, 1);

        // ---- WMMA: 2 (M) x 2 (N) tiles, gate and up share the A fragment ----
        v16bf a[2], bg[2], bu[2];
        #pragma unroll
        for (int i = 0; i < 2; ++i)
            a[i] = load_frag(sA, wm * 32 + i * 16, lane);
        #pragma unroll
        for (int j = 0; j < 2; ++j) {
            bg[j] = load_frag(sG, wn * 32 + j * 16, lane);
            bu[j] = load_frag(sU, wn * 32 + j * 16, lane);
        }
        #pragma unroll
        for (int i = 0; i < 2; ++i)
            #pragma unroll
            for (int j = 0; j < 2; ++j) {
                accG[i][j] = wmma_bf16(a[i], bg[j], accG[i][j]);
                accU[i][j] = wmma_bf16(a[i], bu[j], accU[i][j]);
            }
    }

    // ---- epilogue: act = gelu(gate) * up, store bf16 ----
    const int r    = lane & 15;
    const int half = (lane >> 4) & 1;
    #pragma unroll
    for (int i = 0; i < 2; ++i) {
        #pragma unroll
        for (int j = 0; j < 2; ++j) {
            #pragma unroll
            for (int e = 0; e < 8; ++e) {
                int row = m0 + wm * 32 + i * 16 + half * 8 + e;
                int col = n0 + wn * 32 + j * 16 + r;
                float g = accG[i][j][e];
                float u = accU[i][j][e];
                act[(size_t)row * DIM_I + col] = f2bf(gelu_tanh(g) * u);
            }
        }
    }
}

// ---------------------------------------------------------------------------
// Stage 2: out += w_e(token) * (act @ W2e), one M-chunk.
//   act [M_CHUNK][I] bf16 | W2 [I][H] f32 | sel/rw [M_CHUNK][2] | out f32
// grid = (M_CHUNK/BM, H/BN), block = 256.
__global__ __launch_bounds__(256) void moe_stage2(
        const unsigned short* __restrict__ act,
        const float* __restrict__ W2,
        const int*   __restrict__ sel,
        const float* __restrict__ rw,
        float* __restrict__ out,
        int expert) {
    __shared__ unsigned short sA[BM * BK];
    __shared__ unsigned short sB[BN * BK];

    const int tid  = threadIdx.x;
    const int lane = tid & 31;
    const int wave = tid >> 5;
    const int wm = wave & 3;
    const int wn = wave >> 2;
    const int m0 = blockIdx.x * BM;
    const int n0 = blockIdx.y * BN;

    v8f acc[2][2] = {};

    const int kw = tid >> 4;
    const int nw = (tid & 15) * 4;

    // async A-tile chunk mapping: 512 x 16B chunks, 2 per thread
    const int c0row = (tid + 0)   >> 2, c0k = ((tid + 0)   & 3) * 8;
    const int c1row = (tid + 256) >> 2, c1k = ((tid + 256) & 3) * 8;
    const unsigned ldsA0 =
        (unsigned)(size_t)(const void*)&sA[c0row * BK + c0k];
    const unsigned ldsA1 =
        (unsigned)(size_t)(const void*)&sA[c1row * BK + c1k];

    for (int k0 = 0; k0 < DIM_I; k0 += BK) {
        __syncthreads();

        // ---- A tile: bf16 copy via async global->LDS (ASYNCcnt) ----
        async_copy_b128(ldsA0,
            act + (size_t)(m0 + c0row) * DIM_I + k0 + c0k);
        async_copy_b128(ldsA1,
            act + (size_t)(m0 + c1row) * DIM_I + k0 + c1k);

        // ---- W2 tile: coalesced float4, transposed bf16 scatter ----
        float4 fb[2];
        #pragma unroll
        for (int p = 0; p < 2; ++p)
            fb[p] = *(const float4*)(W2 + (size_t)(k0 + kw + p * 16) * DIM_H
                                        + n0 + nw);
        #pragma unroll
        for (int p = 0; p < 2; ++p) {
            int k = kw + p * 16;
            sB[(nw + 0) * BK + k] = f2bf(fb[p].x);
            sB[(nw + 1) * BK + k] = f2bf(fb[p].y);
            sB[(nw + 2) * BK + k] = f2bf(fb[p].z);
            sB[(nw + 3) * BK + k] = f2bf(fb[p].w);
        }

        wait_asynccnt0();     // our async LDS writes have landed
        __syncthreads();      // everyone's writes visible

        if (k0 + BK < DIM_I)
            __builtin_prefetch(act + (size_t)(m0 + c0row) * DIM_I + k0 + BK,
                               0, 1);

        v16bf a[2], b[2];
        #pragma unroll
        for (int i = 0; i < 2; ++i)
            a[i] = load_frag(sA, wm * 32 + i * 16, lane);
        #pragma unroll
        for (int j = 0; j < 2; ++j)
            b[j] = load_frag(sB, wn * 32 + j * 16, lane);
        #pragma unroll
        for (int i = 0; i < 2; ++i)
            #pragma unroll
            for (int j = 0; j < 2; ++j)
                acc[i][j] = wmma_bf16(a[i], b[j], acc[i][j]);
    }

    // ---- epilogue: scale by routing weight, accumulate into out ----
    const int r    = lane & 15;
    const int half = (lane >> 4) & 1;
    #pragma unroll
    for (int i = 0; i < 2; ++i) {
        #pragma unroll
        for (int e = 0; e < 8; ++e) {
            int row = m0 + wm * 32 + i * 16 + half * 8 + e;
            int s0 = sel[row * 2 + 0], s1 = sel[row * 2 + 1];
            float w = (s0 == expert ? rw[row * 2 + 0] : 0.f) +
                      (s1 == expert ? rw[row * 2 + 1] : 0.f);
            #pragma unroll
            for (int j = 0; j < 2; ++j) {
                int col = n0 + wn * 32 + j * 16 + r;
                float* o = out + (size_t)row * DIM_H + col;
                *o += w * acc[i][j][e];   // race-free: experts serialized on stream
            }
        }
    }
}

// ---------------------------------------------------------------------------
extern "C" void kernel_launch(void* const* d_in, const int* in_sizes, int n_in,
                              void* d_out, int out_size, void* d_ws, size_t ws_size,
                              hipStream_t stream) {
    // setup_inputs order:
    // 0 selected_experts [B,S,2] i32 | 1 hidden_states [B,S,H] f32
    // 2 routing_weights  [B,S,2] f32 | 3..5 scalars
    // 6 w1 [E,H,I] f32 | 7 wv [E,H,I] f32 | 8 w2 [E,I,H] f32
    const int*   sel = (const int*)d_in[0];
    const float* X   = (const float*)d_in[1];
    const float* rw  = (const float*)d_in[2];
    const float* w1  = (const float*)d_in[6];
    const float* wv  = (const float*)d_in[7];
    const float* w2  = (const float*)d_in[8];
    float* out = (float*)d_out;

    unsigned short* act = (unsigned short*)d_ws;   // 8 MB bf16 chunk

    {   // zero the fp32 output (harness poisons it)
        int n4 = (DIM_M_TOTAL * DIM_H) / 4;
        moe_zero_out<<<(n4 + 255) / 256, 256, 0, stream>>>((float4*)out, n4);
    }

    dim3 block(256);
    dim3 g1(M_CHUNK / BM, DIM_I / BN);   // (8, 64)
    dim3 g2(M_CHUNK / BM, DIM_H / BN);   // (8, 16)

    for (int e = 0; e < NUM_EXPERTS; ++e) {
        const float* W1e = w1 + (size_t)e * DIM_H * DIM_I;
        const float* Wve = wv + (size_t)e * DIM_H * DIM_I;
        const float* W2e = w2 + (size_t)e * DIM_I * DIM_H;
        for (int mc = 0; mc < DIM_M_TOTAL; mc += M_CHUNK) {
            moe_stage1<<<g1, block, 0, stream>>>(
                X + (size_t)mc * DIM_H, W1e, Wve, act);
            moe_stage2<<<g2, block, 0, stream>>>(
                act, W2e, sel + (size_t)mc * 2, rw + (size_t)mc * 2,
                out + (size_t)mc * DIM_H, e);
        }
    }
}
